// WorldStateEncoder_68418829025635
// MI455X (gfx1250) — compile-verified, matrix operand order checked by reading, not run on hardware
//
#include <hip/hip_runtime.h>
#include <hip/hip_bf16.h>

// ---------------------------------------------------------------------------
// WorldStateEncoder on gfx1250 (MI455X).
//  - All GEMMs use v_wmma_f32_16x16x32_f16 (wave32, M=16 batch tile per wave).
//  - Weights converted once per launch to f16, column-major (K padded to 32),
//    into d_ws; they are tiny and stay resident in L2/WGP$.
//  - Attention over <=20 agents is streamed (online softmax); Q held in packed
//    f16 registers, K/V read as ds_load_b128 vectors (f16xf32 -> v_fma_mix).
// ---------------------------------------------------------------------------

typedef __attribute__((ext_vector_type(16))) _Float16 v16h;
typedef __attribute__((ext_vector_type(8)))  _Float16 v8h;
typedef __attribute__((ext_vector_type(8)))  float    v8f;

#define DEV __device__ __forceinline__

// ---------------- weight prep: f32 KxN row-major -> f16 column-major, K padded
__global__ void cvt_w_kernel(const float* __restrict__ W, _Float16* __restrict__ out,
                             int K, int Kpad, int N) {
    int idx = blockIdx.x * 256 + threadIdx.x;
    if (idx >= N * Kpad) return;
    int n = idx / Kpad;
    int k = idx - n * Kpad;
    out[idx] = (k < K) ? (_Float16)W[k * N + n] : (_Float16)0.0f;
}

// ---------------- WMMA fragment load ---------------------------------------
// 16-bit A/B layout (ISA 7.12.2): lane<16 -> row/col = lane, K = {0..7,16..23};
// lane>=16 -> row/col = lane-16, K = {8..15,24..31}.  Two 16B loads per frag.
DEV v16h load_frag(const _Float16* base, int stride, int kOff, int lane) {
    int rc   = lane & 15;
    int koff = kOff + ((lane >> 4) << 3);
    const _Float16* p = base + rc * stride + koff;
    v16h f;
    ((v8h*)&f)[0] = *(const v8h*)(p);
    ((v8h*)&f)[1] = *(const v8h*)(p + 16);
    return f;
}

// ---------------- generic 16xNxK GEMM, D -> LDS f16 -------------------------
template<int RELU>
DEV void gemm_lds(const _Float16* sA, int lda,
                  const _Float16* gB, int ldb,       // column-major: [n*ldb + k]
                  const float* __restrict__ bias,
                  int K, int N,
                  _Float16* sD, int ldd, int lane) {
    const int nl = lane & 15;
    const int mh = (lane >> 4) * 8;
    const int kTiles = K >> 5;
    const int nTiles = N >> 4;
    for (int nt = 0; nt < nTiles; ++nt) {
        v8f acc = {};
        const _Float16* bcol = gB + (nt * 16) * ldb;
        for (int kt = 0; kt < kTiles; ++kt) {
            v16h a  = load_frag(sA,   lda, kt * 32, lane);
            v16h bm = load_frag(bcol, ldb, kt * 32, lane);
            acc = __builtin_amdgcn_wmma_f32_16x16x32_f16(false, a, false, bm,
                                                         (short)0, acc, false, false);
        }
        const float bb = bias[nt * 16 + nl];
#pragma unroll
        for (int i = 0; i < 8; ++i) {
            float v = acc[i] + bb;
            if (RELU) v = fmaxf(v, 0.0f);
            sD[(mh + i) * ldd + nt * 16 + nl] = (_Float16)v;
        }
    }
}

// ---------------- generic 16xNxK GEMM, D -> global f32 ----------------------
DEV void gemm_out(const _Float16* sA, int lda,
                  const _Float16* gB, int ldb,
                  const float* __restrict__ bias,
                  int K, int N,
                  float* __restrict__ gD, int ldg, int lane) {
    const int nl = lane & 15;
    const int mh = (lane >> 4) * 8;
    const int kTiles = K >> 5;
    const int nTiles = N >> 4;
    for (int nt = 0; nt < nTiles; ++nt) {
        v8f acc = {};
        const _Float16* bcol = gB + (nt * 16) * ldb;
        for (int kt = 0; kt < kTiles; ++kt) {
            v16h a  = load_frag(sA,   lda, kt * 32, lane);
            v16h bm = load_frag(bcol, ldb, kt * 32, lane);
            acc = __builtin_amdgcn_wmma_f32_16x16x32_f16(false, a, false, bm,
                                                         (short)0, acc, false, false);
        }
        const float bb = bias[nt * 16 + nl];
#pragma unroll
        for (int i = 0; i < 8; ++i)
            gD[(mh + i) * ldg + nt * 16 + nl] = acc[i] + bb;
    }
}

// ---------------- attention VALU helpers (vector LDS, fma_mix) --------------
DEV float dot32_qk(const v8h* qv, const _Float16* kp) {
    float s = 0.0f;
#pragma unroll
    for (int j = 0; j < 4; ++j) {
        v8h k = *(const v8h*)(kp + 8 * j);
        v8h q = qv[j];
#pragma unroll
        for (int e = 0; e < 8; ++e)
            s = fmaf((float)q[e], (float)k[e], s);
    }
    return s;
}

DEV void ctx_update(float* c, float sc, float pp, const _Float16* vp) {
#pragma unroll
    for (int j = 0; j < 4; ++j) {
        v8h v = *(const v8h*)(vp + 8 * j);
#pragma unroll
        for (int e = 0; e < 8; ++e)
            c[j * 8 + e] = fmaf(c[j * 8 + e], sc, pp * (float)v[e]);
    }
}

struct Params {
    const float* ego;            // (B,5)
    const float* agents;         // (B,20,6)
    const int*   nA;             // (B,)
    const float *b1e, *b2e, *b1a, *b2a, *bq, *bk, *bv, *bo, *bf1, *bf2;
    const _Float16 *w1e, *w2e, *w1a, *w2a, *wq, *wk, *wv, *wo, *wf1, *wf2;
    float* out;                  // (B,256)
};

// ---------------- main kernel: one wave handles 16 batch rows ---------------
__global__ void __launch_bounds__(32)
wse_kernel(Params p) {
    __shared__ alignas(16) _Float16 sComb[16 * 256]; // [ego_f | agg]
    __shared__ alignas(16) _Float16 sF1[16 * 256];
    __shared__ alignas(16) _Float16 sIn[16 * 32];
    __shared__ alignas(16) _Float16 sHid[16 * 64];
    __shared__ alignas(16) _Float16 sAgF[16 * 128];
    __shared__ alignas(16) _Float16 sQ[16 * 128];
    __shared__ alignas(16) _Float16 sKV[16 * 128];
    __shared__ alignas(16) _Float16 sCtx[16 * 128];

    const int lane = threadIdx.x;
    const int row0 = blockIdx.x * 16;

    // ---------------- ego encoder: (16,5) -> (16,128) into sComb[:,0:128]
    for (int i = lane; i < 16 * 32; i += 32) sIn[i] = (_Float16)0.0f;
    for (int i = lane; i < 16 * 5; i += 32) {
        int r = i / 5, c = i - r * 5;
        sIn[r * 32 + c] = (_Float16)p.ego[(row0 + r) * 5 + c];
    }
    __syncthreads();
    gemm_lds<1>(sIn, 32, p.w1e, 32, p.b1e, 32, 64, sHid, 64, lane);
    gemm_lds<1>(sHid, 64, p.w2e, 64, p.b2e, 64, 128, sComb, 256, lane);
    // Q projection: (16,128) @ Wq
    gemm_lds<0>(sComb, 256, p.wq, 128, p.bq, 128, 128, sQ, 128, lane);
    __syncthreads();

    // ---------------- streamed attention (online softmax over 20 agents)
    const int b  = lane & 15;     // batch row within tile
    const int h0 = lane >> 4;     // this lane owns heads h0 and h0+2
    const int nag = p.nA[row0 + b];

    // hoist Q into packed-f16 registers (loop-invariant across agents)
    v8h qv0[4], qv1[4];
#pragma unroll
    for (int j = 0; j < 4; ++j) {
        qv0[j] = *(const v8h*)(sQ + b * 128 + h0 * 32 + 8 * j);
        qv1[j] = *(const v8h*)(sQ + b * 128 + (h0 + 2) * 32 + 8 * j);
    }

    float m0 = -1e30f, m1 = -1e30f, l0 = 0.0f, l1 = 0.0f;
    float c0[32], c1[32];
#pragma unroll
    for (int d = 0; d < 32; ++d) { c0[d] = 0.0f; c1[d] = 0.0f; }

    for (int a = 0; a < 20; ++a) {
        // agent encoder for agent a: (16,6) -> (16,128)
        for (int i = lane; i < 16 * 32; i += 32) sIn[i] = (_Float16)0.0f;
        for (int i = lane; i < 16 * 6; i += 32) {
            int r = i / 6, c = i - r * 6;
            sIn[r * 32 + c] = (_Float16)p.agents[((row0 + r) * 20 + a) * 6 + c];
        }
        __syncthreads();
        gemm_lds<1>(sIn, 32, p.w1a, 32, p.b1a, 32, 64, sHid, 64, lane);
        gemm_lds<1>(sHid, 64, p.w2a, 64, p.b2a, 64, 128, sAgF, 128, lane);

        // K projection -> sKV, then per-(row,head) scores
        gemm_lds<0>(sAgF, 128, p.wk, 128, p.bk, 128, 128, sKV, 128, lane);
        __syncthreads();
        float s0 = dot32_qk(qv0, sKV + b * 128 + h0 * 32);
        float s1 = dot32_qk(qv1, sKV + b * 128 + (h0 + 2) * 32);
        const float scale = 0.17677669529663687f;          // 1/sqrt(32)
        s0 *= scale; s1 *= scale;
        const bool valid = a < nag;
        float sm0 = valid ? s0 : -1e30f;
        float sm1 = valid ? s1 : -1e30f;
        float nm0 = fmaxf(m0, sm0), nm1 = fmaxf(m1, sm1);
        float sc0 = __expf(m0 - nm0), sc1 = __expf(m1 - nm1);
        float p0 = valid ? __expf(sm0 - nm0) : 0.0f;
        float p1 = valid ? __expf(sm1 - nm1) : 0.0f;
        m0 = nm0; m1 = nm1;

        // V projection -> sKV (reuse), then online context update
        gemm_lds<0>(sAgF, 128, p.wv, 128, p.bv, 128, 128, sKV, 128, lane);
        __syncthreads();
        ctx_update(c0, sc0, p0, sKV + b * 128 + h0 * 32);
        ctx_update(c1, sc1, p1, sKV + b * 128 + (h0 + 2) * 32);
        l0 = l0 * sc0 + p0;
        l1 = l1 * sc1 + p1;
        __syncthreads();
    }

    // normalize context, store f16, project with Wo into sComb[:,128:256]
    const float inv0 = 1.0f / l0, inv1 = 1.0f / l1;
#pragma unroll
    for (int d = 0; d < 32; ++d) {
        sCtx[b * 128 + h0 * 32 + d]       = (_Float16)(c0[d] * inv0);
        sCtx[b * 128 + (h0 + 2) * 32 + d] = (_Float16)(c1[d] * inv1);
    }
    __syncthreads();
    gemm_lds<0>(sCtx, 128, p.wo, 128, p.bo, 128, 128, sComb + 128, 256, lane);

    // fusion MLP: (16,256) -> relu -> (16,256) -> global f32
    gemm_lds<1>(sComb, 256, p.wf1, 256, p.bf1, 256, 256, sF1, 256, lane);
    gemm_out(sF1, 256, p.wf2, 256, p.bf2, 256, 256, p.out + (size_t)row0 * 256, 256, lane);
}

// ---------------------------------------------------------------------------
extern "C" void kernel_launch(void* const* d_in, const int* in_sizes, int n_in,
                              void* d_out, int out_size, void* d_ws, size_t ws_size,
                              hipStream_t stream) {
    const int B = in_sizes[0] / 5;

    _Float16* ws = (_Float16*)d_ws;
    // workspace layout (halves); total = 217088 halves = 434,176 bytes
    size_t off = 0;
    _Float16* w1e = ws + off; off += 64  * 32;   // K=5  ->Kpad32, N=64
    _Float16* w2e = ws + off; off += 128 * 64;   // K=64,         N=128
    _Float16* w1a = ws + off; off += 64  * 32;   // K=6  ->Kpad32, N=64
    _Float16* w2a = ws + off; off += 128 * 64;
    _Float16* wq  = ws + off; off += 128 * 128;
    _Float16* wk  = ws + off; off += 128 * 128;
    _Float16* wv  = ws + off; off += 128 * 128;
    _Float16* wo  = ws + off; off += 128 * 128;
    _Float16* wf1 = ws + off; off += 256 * 256;
    _Float16* wf2 = ws + off; off += 256 * 256;

    struct WSpec { int in_idx; _Float16* dst; int K, Kpad, N; };
    const WSpec specs[10] = {
        { 3, w1e,   5,  32,  64}, { 5, w2e,  64,  64, 128},
        { 7, w1a,   6,  32,  64}, { 9, w2a,  64,  64, 128},
        {11, wq,  128, 128, 128}, {13, wk,  128, 128, 128},
        {15, wv,  128, 128, 128}, {17, wo,  128, 128, 128},
        {19, wf1, 256, 256, 256}, {21, wf2, 256, 256, 256},
    };
    for (int i = 0; i < 10; ++i) {
        int n = specs[i].N * specs[i].Kpad;
        cvt_w_kernel<<<(n + 255) / 256, 256, 0, stream>>>(
            (const float*)d_in[specs[i].in_idx], specs[i].dst,
            specs[i].K, specs[i].Kpad, specs[i].N);
    }

    Params p;
    p.ego    = (const float*)d_in[0];
    p.agents = (const float*)d_in[1];
    p.nA     = (const int*)  d_in[2];
    p.b1e = (const float*)d_in[4];  p.b2e = (const float*)d_in[6];
    p.b1a = (const float*)d_in[8];  p.b2a = (const float*)d_in[10];
    p.bq  = (const float*)d_in[12]; p.bk  = (const float*)d_in[14];
    p.bv  = (const float*)d_in[16]; p.bo  = (const float*)d_in[18];
    p.bf1 = (const float*)d_in[20]; p.bf2 = (const float*)d_in[22];
    p.w1e = w1e; p.w2e = w2e; p.w1a = w1a; p.w2a = w2a;
    p.wq = wq; p.wk = wk; p.wv = wv; p.wo = wo; p.wf1 = wf1; p.wf2 = wf2;
    p.out = (float*)d_out;

    wse_kernel<<<B / 16, 32, 0, stream>>>(p);
}
